// MultiHeadSelfAttentionWithRelativeEncoding_6846177869980
// MI455X (gfx1250) — compile-verified, matrix operand order checked by reading, not run on hardware
//
#include <hip/hip_runtime.h>
#include <hip/hip_bf16.h>

typedef __attribute__((ext_vector_type(16))) __bf16 v16bf;
typedef __attribute__((ext_vector_type(8)))  float  v8f;
typedef __attribute__((__vector_size__(16))) int    v4i;

#if __has_builtin(__builtin_amdgcn_global_load_async_to_lds_b128) && \
    __has_builtin(__builtin_amdgcn_s_wait_asynccnt)
#define HAVE_ASYNC_LDS 1
#define AS_GLOBAL __attribute__((address_space(1)))
#define AS_LDS    __attribute__((address_space(3)))
#endif

// ---------------- problem constants ----------------
constexpr int NN    = 768;
constexpr int INDIM = 128;
constexpr int HID   = 256;
constexpr int HEADS = 8;
constexpr int HD    = 32;
constexpr float RSCALE = 0.17677669529663687f;   // 1/sqrt(32)

// ---------------- workspace layout (bytes) ----------------
constexpr size_t QF_OFF = 0;                         // 768*256*4
constexpr size_t KF_OFF = 786432;
constexpr size_t VF_OFF = 1572864;
constexpr size_t S_OFF  = 2359296;                   // 8*768*768*4 = 18874368
constexpr size_t O1_OFF = 21233664;                  // 768*256*4
constexpr size_t BF_OFF = 22020096;                  // bf16 arena (16B aligned)

// bf16 arena element offsets
constexpr int XB  = 0;        // x bf16: 768*128
constexpr int WQT = 98304;    // Wq^T bf16: 256*128
constexpr int WKT = 131072;
constexpr int WVT = 163840;
constexpr int W2T = 196608;   // W2^T: 256*256
constexpr int W3T = 262144;   // W3^T: 32*256
constexpr int WOT = 270336;   // Wo^T: 256*256
constexpr int BF_TOTAL = 335872;

// ---------------- WMMA helpers ----------------
__device__ inline int frag_k(int e, int hi) {
  // 16-bit A-matrix 16x32 lane layout (ISA 7.12.2): lane<16 -> K {0..7,16..23},
  // lane>=16 -> K {8..15,24..31}; two contiguous 16B chunks -> b128-friendly.
  return hi * 8 + e + (e >= 8 ? 8 : 0);
}

template <typename T>
__device__ inline v16bf load_frag(const T* p, int row0, int k0, int ld) {
  int lane = threadIdx.x & 31;
  int m = lane & 15, hi = lane >> 4;
  const T* b = p + (size_t)(row0 + m) * ld + k0;
  v16bf f;
#pragma unroll
  for (int e = 0; e < 16; ++e) f[e] = (__bf16)b[frag_k(e, hi)];
  return f;
}

__device__ inline v8f wmma_bf16(v16bf a, v16bf b, v8f c) {
  return __builtin_amdgcn_wmma_f32_16x16x32_bf16(false, a, false, b, (short)0, c,
                                                 false, false);
}

// ---------------- kernel 0: bf16 weight transposes ----------------
__global__ __launch_bounds__(256) void prep_kernel(
    const float* x, const float* Wq, const float* Wk, const float* Wv,
    const float* W2, const float* W3, const float* Wo, __bf16* arena) {
  int idx = blockIdx.x * blockDim.x + threadIdx.x;
  if (idx >= BF_TOTAL) return;
  if (idx < 98304) { arena[XB + idx] = (__bf16)x[idx]; return; }
  int r = idx - 98304;
  if (r < 3 * 32768) {
    int m = r / 32768, q = r % 32768;
    int nrow = q >> 7, k = q & 127;
    const float* W = (m == 0) ? Wq : (m == 1) ? Wk : Wv;
    arena[WQT + r] = (__bf16)W[k * 256 + nrow];
    return;
  }
  r -= 3 * 32768;
  if (r < 65536) { int n = r >> 8, k = r & 255; arena[W2T + r] = (__bf16)W2[k * 256 + n]; return; }
  r -= 65536;
  if (r < 8192)  { int d = r >> 8, k = r & 255; arena[W3T + r] = (__bf16)W3[k * 32 + d]; return; }
  r -= 8192;
  { int n = r >> 8, k = r & 255; arena[WOT + r] = (__bf16)Wo[k * 256 + n]; }
}

// ---------------- kernel 1: QKV projections (WMMA bf16) ----------------
__global__ __launch_bounds__(32) void qkv_kernel(
    const __bf16* arena, const float* bq, const float* bk, const float* bv,
    float* Qf, float* Kf, float* Vf) {
  int it  = blockIdx.x;          // row tile 0..47
  int mat = blockIdx.y >> 4;     // 0=Q 1=K 2=V
  int ct  = blockIdx.y & 15;     // col tile 0..15
  const __bf16* xb = arena + XB;
  const __bf16* Wt = arena + ((mat == 0) ? WQT : (mat == 1) ? WKT : WVT);
  const float* bias = (mat == 0) ? bq : (mat == 1) ? bk : bv;
  float* dst = (mat == 0) ? Qf : (mat == 1) ? Kf : Vf;
  v8f acc = {};
  for (int k0 = 0; k0 < INDIM; k0 += 32) {
    v16bf a = load_frag(xb, it * 16, k0, INDIM);
    v16bf b = load_frag(Wt, ct * 16, k0, INDIM);
    acc = wmma_bf16(a, b, acc);
  }
  int lane = threadIdx.x & 31, hi = lane >> 4;
  int ncol = ct * 16 + (lane & 15);
  float bb = bias[ncol];
#pragma unroll
  for (int r = 0; r < 8; ++r)
    dst[(size_t)(it * 16 + r + hi * 8) * HID + ncol] = acc[r] + bb;
}

// ---------------- kernel 2: fused pair-MLP + scores ----------------
// One workgroup owns a 16x16 tile of (i,j) pairs -> M = 256 rows.
// Entirely in LDS (CDNA5: 320KB per WGP): h1 (bf16 256x280) + h2 (bf16 256x280).
constexpr int H1_LD    = 280;                       // bf16 elems; 560B row: 16B aligned, bank-spread
constexpr int H1_BYTES = 256 * H1_LD * 2;           // 143360
constexpr int SMEM_BYTES = 2 * H1_BYTES;            // 286720 <= 320KB
constexpr int ENC_LD = 33;                          // f32 elems
constexpr int QK_LD  = 260;                         // f32 elems; 1040B row: 16B aligned
constexpr int QLDS_OFF = 36864;                     // bytes within region 0
constexpr int KLDS_OFF = 57344;                     // 16x260x4=16640 per tile; fits

__global__ __launch_bounds__(256) void relenc_kernel(
    const float* pos, const float* W1, const float* b1,
    const float* b2, const float* b3,
    const __bf16* arena, const float* Qf, const float* Kf, float* S) {
  extern __shared__ __align__(16) char smem[];
  __bf16* h1 = (__bf16*)smem;
  __bf16* h2 = (__bf16*)(smem + H1_BYTES);
  float* enc = (float*)smem;                   // reuses h1 region after phase B
  float* qld = (float*)(smem + QLDS_OFF);
  float* kld = (float*)(smem + KLDS_OFF);

  int I = blockIdx.x, J = blockIdx.y;
  int t = threadIdx.x;
  int lane = t & 31, wave = t >> 5, hi = lane >> 4;

  const __bf16* W2t = arena + W2T;
  const __bf16* W3t = arena + W3T;

  // Warm L2/WGP$ with the W2^T block this workgroup is about to stream.
  __builtin_prefetch(((const char*)W2t) + t * 512, 0, 2);
  __builtin_prefetch(((const char*)W2t) + t * 512 + 256, 0, 2);

  // ---- Phase A: h1 = relu(rel @ W1 + b1), rel = pos[j] - pos[i]  (VALU f32, K=3)
  {
    int i = t >> 4, j = t & 15;
    int gi = I * 16 + i, gj = J * 16 + j;
    float r0 = pos[gj * 3 + 0] - pos[gi * 3 + 0];
    float r1 = pos[gj * 3 + 1] - pos[gi * 3 + 1];
    float r2 = pos[gj * 3 + 2] - pos[gi * 3 + 2];
#pragma unroll 4
    for (int n = 0; n < HID; ++n) {
      float v = fmaf(r0, W1[n], fmaf(r1, W1[256 + n], fmaf(r2, W1[512 + n], b1[n])));
      h1[t * H1_LD + n] = (__bf16)fmaxf(v, 0.f);
    }
  }
  __syncthreads();

  // ---- Phase B: h2 = relu(h1 @ W2 + b2)   M=256 K=256 N=256, bf16 WMMA
  // A-fragments register-cached per M-tile; reused across all 16 N-tiles.
  for (int mt = wave * 2; mt < wave * 2 + 2; ++mt) {
    v16bf aF[8];
#pragma unroll
    for (int kk = 0; kk < 8; ++kk) aF[kk] = load_frag(h1, mt * 16, kk * 32, H1_LD);
    for (int nt = 0; nt < 16; ++nt) {
      v8f acc = {};
#pragma unroll
      for (int kk = 0; kk < 8; ++kk)
        acc = wmma_bf16(aF[kk], load_frag(W2t, nt * 16, kk * 32, 256), acc);
      int ncol = nt * 16 + (lane & 15);
      float bias = b2[ncol];
#pragma unroll
      for (int r = 0; r < 8; ++r) {
        float v = fmaxf(acc[r] + bias, 0.f);
        h2[(mt * 16 + r + hi * 8) * H1_LD + ncol] = (__bf16)v;
      }
    }
  }
  __syncthreads();

  // ---- Stage Q (I-tile rows) and K (J-tile rows) into LDS.
  // Async DMA (ASYNCcnt) overlaps the phase-C WMMAs; h1 region is dead here.
#ifdef HAVE_ASYNC_LDS
#pragma unroll
  for (int rr = 0; rr < 4; ++rr) {
    int chunk = t + rr * 256;              // 1024 chunks of 4 floats per matrix
    int row = chunk >> 6, c4 = (chunk & 63) * 4;
    __builtin_amdgcn_global_load_async_to_lds_b128(
        (AS_GLOBAL v4i*)(Qf + (size_t)(I * 16 + row) * HID + c4),
        (AS_LDS    v4i*)(qld + row * QK_LD + c4), 0, 0);
    __builtin_amdgcn_global_load_async_to_lds_b128(
        (AS_GLOBAL v4i*)(Kf + (size_t)(J * 16 + row) * HID + c4),
        (AS_LDS    v4i*)(kld + row * QK_LD + c4), 0, 0);
  }
#endif

  // ---- Phase C: enc = h2 @ W3 + b3  (M=256 K=256 N=32)
  for (int mt = wave * 2; mt < wave * 2 + 2; ++mt) {
    v16bf aF[8];
#pragma unroll
    for (int kk = 0; kk < 8; ++kk) aF[kk] = load_frag(h2, mt * 16, kk * 32, H1_LD);
#pragma unroll
    for (int nt = 0; nt < 2; ++nt) {
      v8f acc = {};
#pragma unroll
      for (int kk = 0; kk < 8; ++kk)
        acc = wmma_bf16(aF[kk], load_frag(W3t, nt * 16, kk * 32, 256), acc);
      int ncol = nt * 16 + (lane & 15);
      float bias = b3[ncol];
#pragma unroll
      for (int r = 0; r < 8; ++r)
        enc[(mt * 16 + r + hi * 8) * ENC_LD + ncol] = acc[r] + bias;
    }
  }

#ifdef HAVE_ASYNC_LDS
  __builtin_amdgcn_s_wait_asynccnt(0);
#else
#pragma unroll
  for (int rr = 0; rr < 4; ++rr) {
    int chunk = t + rr * 256;
    int row = chunk >> 6, c4 = (chunk & 63) * 4;
    *(float4*)(qld + row * QK_LD + c4) =
        *(const float4*)(Qf + (size_t)(I * 16 + row) * HID + c4);
    *(float4*)(kld + row * QK_LD + c4) =
        *(const float4*)(Kf + (size_t)(J * 16 + row) * HID + c4);
  }
#endif
  __syncthreads();

  // ---- Phase D: S[h,i,j] = (Q.K + Q.enc) / sqrt(d)   (f32 VALU)
  {
    int i = t >> 4, j = t & 15;
    int gi = I * 16 + i, gj = J * 16 + j;
#pragma unroll
    for (int h = 0; h < HEADS; ++h) {
      float s = 0.f;
#pragma unroll
      for (int d = 0; d < HD; ++d) {
        float q = qld[i * QK_LD + h * HD + d];
        s = fmaf(q, kld[j * QK_LD + h * HD + d] + enc[t * ENC_LD + d], s);
      }
      S[((size_t)h * NN + gi) * NN + gj] = s * RSCALE;
    }
  }
}

// ---------------- kernel 3: softmax over j (wave per (h,i) row) ----------------
__global__ __launch_bounds__(256) void softmax_kernel(float* S) {
  int gi = blockIdx.x;
  int lane = threadIdx.x & 31, h = threadIdx.x >> 5;
  float* row = S + ((size_t)h * NN + gi) * NN;
  float vals[24];
  float mx = -3.4e38f;
#pragma unroll
  for (int l = 0; l < 24; ++l) { vals[l] = row[lane + l * 32]; mx = fmaxf(mx, vals[l]); }
#pragma unroll
  for (int off = 16; off > 0; off >>= 1) mx = fmaxf(mx, __shfl_xor(mx, off, 32));
  float sum = 0.f;
#pragma unroll
  for (int l = 0; l < 24; ++l) { vals[l] = __expf(vals[l] - mx); sum += vals[l]; }
#pragma unroll
  for (int off = 16; off > 0; off >>= 1) sum += __shfl_xor(sum, off, 32);
  float inv = 1.f / sum;
#pragma unroll
  for (int l = 0; l < 24; ++l) row[lane + l * 32] = vals[l] * inv;
}

// ---------------- kernel 4: O1[i, h*32+d] = sum_j P[h,i,j] * V[j, h*32+d] ----------------
__global__ __launch_bounds__(64) void av_kernel(const float* S, const float* Vf, float* O1) {
  int it = blockIdx.x;             // 0..47
  int h  = blockIdx.y;             // 0..7
  int lane = threadIdx.x & 31, nt = threadIdx.x >> 5;   // 2 N-tiles per block
  int m = lane & 15, hi = lane >> 4, nc = lane & 15;
  const float* Arow = S + ((size_t)h * NN + it * 16 + m) * NN;
  v8f acc = {};
  for (int k0 = 0; k0 < NN; k0 += 32) {
    v16bf a, b;
#pragma unroll
    for (int e = 0; e < 16; ++e) {
      int k = k0 + frag_k(e, hi);
      a[e] = (__bf16)Arow[k];
      b[e] = (__bf16)Vf[(size_t)k * HID + h * HD + nt * 16 + nc];
    }
    acc = wmma_bf16(a, b, acc);
  }
#pragma unroll
  for (int r = 0; r < 8; ++r)
    O1[(size_t)(it * 16 + r + hi * 8) * HID + h * HD + nt * 16 + nc] = acc[r];
}

// ---------------- kernel 5: out = O1 @ Wo + bo ----------------
__global__ __launch_bounds__(32) void outproj_kernel(
    const float* O1, const __bf16* arena, const float* bo, float* out) {
  int it = blockIdx.x, nt = blockIdx.y;
  int lane = threadIdx.x & 31, hi = lane >> 4, nc = lane & 15;
  const __bf16* Wot = arena + WOT;
  v8f acc = {};
#pragma unroll
  for (int kk = 0; kk < 8; ++kk) {
    v16bf a = load_frag(O1, it * 16, kk * 32, HID);    // f32 -> bf16 on the fly
    v16bf b = load_frag(Wot, nt * 16, kk * 32, HID);
    acc = wmma_bf16(a, b, acc);
  }
  int ncol = nt * 16 + nc;
  float bias = bo[ncol];
#pragma unroll
  for (int r = 0; r < 8; ++r)
    out[(size_t)(it * 16 + r + hi * 8) * HID + ncol] = acc[r] + bias;
}

// ---------------- launch ----------------
extern "C" void kernel_launch(void* const* d_in, const int* in_sizes, int n_in,
                              void* d_out, int out_size, void* d_ws, size_t ws_size,
                              hipStream_t stream) {
  (void)in_sizes; (void)n_in; (void)out_size; (void)ws_size;
  const float* x   = (const float*)d_in[0];
  const float* pos = (const float*)d_in[1];
  const float* Wq  = (const float*)d_in[2];  const float* bq = (const float*)d_in[3];
  const float* Wk  = (const float*)d_in[4];  const float* bk = (const float*)d_in[5];
  const float* Wv  = (const float*)d_in[6];  const float* bv = (const float*)d_in[7];
  const float* Wo  = (const float*)d_in[8];  const float* bo = (const float*)d_in[9];
  const float* W1  = (const float*)d_in[10]; const float* b1 = (const float*)d_in[11];
  const float* W2  = (const float*)d_in[12]; const float* b2 = (const float*)d_in[13];
  const float* W3  = (const float*)d_in[14]; const float* b3 = (const float*)d_in[15];

  char* ws = (char*)d_ws;
  float*  Qf    = (float*)(ws + QF_OFF);
  float*  Kf    = (float*)(ws + KF_OFF);
  float*  Vf    = (float*)(ws + VF_OFF);
  float*  S     = (float*)(ws + S_OFF);
  float*  O1    = (float*)(ws + O1_OFF);
  __bf16* arena = (__bf16*)(ws + BF_OFF);
  float*  out   = (float*)d_out;

  (void)hipFuncSetAttribute((const void*)relenc_kernel,
                            hipFuncAttributeMaxDynamicSharedMemorySize, SMEM_BYTES);

  prep_kernel<<<(BF_TOTAL + 255) / 256, 256, 0, stream>>>(x, Wq, Wk, Wv, W2, W3, Wo, arena);
  qkv_kernel<<<dim3(48, 48), 32, 0, stream>>>(arena, bq, bk, bv, Qf, Kf, Vf);
  relenc_kernel<<<dim3(48, 48), 256, SMEM_BYTES, stream>>>(pos, W1, b1, b2, b3,
                                                           arena, Qf, Kf, S);
  softmax_kernel<<<NN, 256, 0, stream>>>(S);
  av_kernel<<<dim3(48, 8), 64, 0, stream>>>(S, Vf, O1);
  outproj_kernel<<<dim3(48, 16), 32, 0, stream>>>(O1, arena, bo, out);
}
